// Codebook_10831907520521
// MI455X (gfx1250) — matrix operand order repaired and checked by reading, not compiled, not run on hardware
//
#include <hip/hip_runtime.h>

#define LATENT_DIM 256
#define NUM_VECTORS 1024
#define M_TILE 64
#define THREADS 128
#define N_ROWS_TOTAL (32 * 64 * 64) /* 131072 */
#define HW 4096
#define BETA 0.25f
#define ROW_PAD (LATENT_DIM + 8) /* bf16 elems per LDS row (16B-aligned stride) */
#define NCHUNKS (NUM_VECTORS / 16) /* 64 column chunks */
#define CHUNK_ELEMS 4096           /* bf16 elems per packed B chunk (8 KB) */

typedef __attribute__((ext_vector_type(16))) __bf16 bf16x16;
typedef __attribute__((ext_vector_type(8))) float f32x8;
typedef __attribute__((ext_vector_type(4))) unsigned int u32x4;

union FragU {
    u32x4 u[2];
    bf16x16 v;
};

static __device__ inline unsigned short f2bf(float f) {
    __bf16 b = (__bf16)f; // RNE convert on gfx1250
    return __builtin_bit_cast(unsigned short, b);
}

// ---------------------------------------------------------------------------
// Tensor Data Mover path (gfx1250): DMA one 8 KB packed-B chunk into LDS.
// ---------------------------------------------------------------------------
#if defined(__has_builtin)
#if __has_builtin(__builtin_amdgcn_tensor_load_to_lds) && \
    __has_builtin(__builtin_amdgcn_s_wait_tensorcnt)
#define HAS_TDM 1
#endif
#endif
#ifndef HAS_TDM
#define HAS_TDM 0
#endif

#if HAS_TDM
typedef __attribute__((ext_vector_type(4))) unsigned int v4u;
typedef __attribute__((ext_vector_type(8))) int v8i;
typedef __attribute__((ext_vector_type(4))) int v4i;

// D# per CDNA5 ISA ch.8: 1 row x 1024 elements of data_size=8B (= 8 KB tile).
static __device__ inline void tdm_load_chunk(const unsigned short* gsrc,
                                             unsigned ldsByteAddr) {
    const unsigned long long ga = (unsigned long long)(size_t)gsrc;
    v4u g0;
    g0[0] = 1u;                   // count=1 (valid), is_restore=0, gather_mode=0
    g0[1] = ldsByteAddr;          // lds_addr[31:0]
    g0[2] = (unsigned)ga;         // global_addr[31:0]
    g0[3] = (unsigned)((ga >> 32) & 0x1FFFFFFu) | (2u << 30); // addr[56:32], type=2
    v8i g1;
    g1[0] = (int)(3u << 16);      // workgroup_mask=0 (no cluster), data_size=3 (8B)
    g1[1] = (int)(1024u << 16);   // tensor_dim0 = 1024  (bits 79:48, low half)
    g1[2] = (int)(1u << 16);      // tensor_dim0 hi=0; tensor_dim1 = 1 (bits 111:80)
    g1[3] = (int)(1024u << 16);   // tensor_dim1 hi=0; tile_dim0 = 1024
    g1[4] = 1;                    // tile_dim1 = 1, tile_dim2 = 0
    g1[5] = 1024;                 // tensor_dim0_stride = 1024
    g1[6] = (int)(1024u << 16);   // tensor_dim1_stride = 1024 (unused, 1 row)
    g1[7] = 0;
    const v4i z4 = {0, 0, 0, 0};
#if __clang_major__ >= 23
    const v8i z8 = {0, 0, 0, 0, 0, 0, 0, 0};
    __builtin_amdgcn_tensor_load_to_lds(g0, g1, z4, z4, z8, 0);
#else
    __builtin_amdgcn_tensor_load_to_lds(g0, g1, z4, z4, 0);
#endif
}
#endif

// ---------------------------------------------------------------------------
// Prep: emb(f32) -> bf16 pre-swizzled into WMMA B-fragment layout, plus ||e||^2.
// packedB[((nc*8 + kc)*32 + lane)*16 + e] = bf16( emb[nc*16+(lane&15)][kc*32+k] ), k = table order of (lane>>4, e)
// so a B fragment for (nc,kc) is 32 contiguous bytes per lane.
// ---------------------------------------------------------------------------
__global__ __launch_bounds__(256) void vq_prep(const float* __restrict__ emb,
                                               unsigned short* __restrict__ packedB,
                                               float* __restrict__ embNorm) {
    const int v = blockIdx.x;  // codebook vector index 0..1023
    const int t = threadIdx.x; // dim index 0..255
    float val = emb[v * LATENT_DIM + t];
    packedB[(((v >> 4) * 8 + (t >> 5)) * 32 + ((t >> 3) & 1) * 16 + (v & 15)) * 16 + (t & 7) + ((t >> 4) & 1) * 8] = f2bf(val);

    __shared__ float red[256];
    red[t] = val * val;
    __syncthreads();
    for (int s = 128; s > 0; s >>= 1) {
        if (t < s) red[t] += red[t + s];
        __syncthreads();
    }
    if (t == 0) embNorm[v] = red[0];
}

// ---------------------------------------------------------------------------
// Main: per 64-row tile: distances via bf16 WMMA, argmin, gather, partial loss.
// ---------------------------------------------------------------------------
__global__ __launch_bounds__(THREADS) void vq_main(const float* __restrict__ x,
                                                   const float* __restrict__ emb,
                                                   const unsigned short* __restrict__ packedB,
                                                   const float* __restrict__ embNorm,
                                                   float* __restrict__ outXq,
                                                   float* __restrict__ outIdx,
                                                   float* __restrict__ lossPartial) {
    __shared__ unsigned short ldsA[M_TILE * ROW_PAD]; // 64*264*2 = 33792 B
    __shared__ unsigned short ldsB[2][CHUNK_ELEMS];   // 2 x 8192 B (double buffer)
    __shared__ float embNormLds[NUM_VECTORS];         // 4096 B
    __shared__ int idxTile[M_TILE];
    __shared__ float redBuf[THREADS];

    const int t = threadIdx.x;
    const int lane = t & 31;
    const int waveId = t >> 5;
    const int rowBlock = blockIdx.x * M_TILE; // global row base (row = b*4096 + hw)
    const int bIdx = rowBlock >> 12;          // batch index (tile never crosses batch)
    const int hw0 = rowBlock & 4095;

#if HAS_TDM
    const unsigned ldsBBase = (unsigned)(size_t)(&ldsB[0][0]);
    // Kick off the DMA of chunk 0 immediately; it overlaps the A staging below.
    if (waveId == 0) tdm_load_chunk(packedB, ldsBBase);
#endif

    // ---- Phase 1: stage x tile (f32 -> bf16) into LDS; stage ||e||^2 ----
    {
        const int r = t & 63;
        const int ch = t >> 6; // 0/1: splits C dimension
        const float* xrow = x + (size_t)bIdx * LATENT_DIM * HW + hw0 + r;
        for (int cc = 0; cc < LATENT_DIM; cc += 2) {
            const int c = cc + ch;
            ldsA[r * ROW_PAD + c] = f2bf(xrow[(size_t)c * HW]); // coalesced over r
        }
        for (int i = 0; i < NUM_VECTORS / THREADS; ++i)
            embNormLds[i * THREADS + t] = embNorm[i * THREADS + t];
    }
    __syncthreads();

    // ---- hoist all A fragments for this wave's 16 rows into registers ----
    // A layout: lane -> row (lane&15); lane[4] selects K-half; elems 0..7 =
    // K[h..h+7], elems 8..15 = K[16+h..23+h] within each 32-wide K chunk.
    const int mLocal = waveId * 16 + (lane & 15);
    const int khalf = (lane >> 4) * 8;
    const unsigned aBase = (unsigned)mLocal * ROW_PAD + khalf;
    FragU aF[8];
#pragma unroll
    for (int kc = 0; kc < 8; ++kc) {
        aF[kc].u[0] = *(const u32x4*)(ldsA + aBase + kc * 32);
        aF[kc].u[1] = *(const u32x4*)(ldsA + aBase + kc * 32 + 16);
    }

    // ---- Phase 2: distance GEMM + running argmin ----
    // C-tile layout: lanes 0-15 -> rows j (j=0..7), lanes 16-31 -> rows 8+j;
    // every accumulator element of a lane shares column n = nc*16 + (lane&15).
    float minV[8];
    int minI[8];
#pragma unroll
    for (int j = 0; j < 8; ++j) {
        minV[j] = 3.4e38f;
        minI[j] = 0;
    }
    const int myCol = lane & 15;

    for (int nc = 0; nc < NCHUNKS; ++nc) {
#if HAS_TDM
        const int buf = nc & 1;
        if (waveId == 0) {
            if (nc + 1 < NCHUNKS) {
                // DMA next chunk into the other buffer, then wait for current.
                tdm_load_chunk(packedB + (size_t)(nc + 1) * CHUNK_ELEMS,
                               ldsBBase + (unsigned)((nc + 1) & 1) * (CHUNK_ELEMS * 2));
                __builtin_amdgcn_s_wait_tensorcnt(1); // in-order: chunk nc done
            } else {
                __builtin_amdgcn_s_wait_tensorcnt(0);
            }
        }
        __syncthreads(); // B(nc) visible to all waves
        const unsigned short* Bbuf = &ldsB[buf][0];
#else
        __syncthreads(); // previous chunk's ldsB reads complete
        {
            const u32x4* src = (const u32x4*)(packedB + nc * CHUNK_ELEMS);
            u32x4* dst = (u32x4*)&ldsB[0][0];
#pragma unroll
            for (int i = 0; i < 4; ++i) dst[t + i * THREADS] = src[t + i * THREADS];
            if (nc + 1 < NCHUNKS)
                __builtin_prefetch(packedB + (nc + 1) * CHUNK_ELEMS + t * 16, 0, 1);
        }
        __syncthreads();
        const unsigned short* Bbuf = &ldsB[0][0];
#endif

        f32x8 acc = {};
#pragma unroll
        for (int kc = 0; kc < 8; ++kc) {
            FragU bb; // B fragment: 32 contiguous bytes per lane (pre-swizzled)
            const u32x4* pb = (const u32x4*)(Bbuf + (kc * 32 + lane) * 16);
            bb.u[0] = pb[0];
            bb.u[1] = pb[1];
            acc = __builtin_amdgcn_wmma_f32_16x16x32_bf16(false, aF[kc].v, false, bb.v,
                                                          (short)0, acc, false, false);
        }

        const float en = embNormLds[nc * 16 + myCol];
        const int colIdx = nc * 16 + myCol;
#pragma unroll
        for (int j = 0; j < 8; ++j) {
            const float s = en - 2.0f * acc[j]; // ||x||^2 dropped: row-constant
            if (s < minV[j]) { // strict '<' keeps earliest index within lane
                minV[j] = s;
                minI[j] = colIdx;
            }
        }
#if HAS_TDM
        __syncthreads(); // all waves done with buf before it is DMA-overwritten
#endif
    }

    // ---- reduce argmin across the 16 lanes sharing each row group ----
#pragma unroll
    for (int j = 0; j < 8; ++j) {
#pragma unroll
        for (int m = 1; m <= 8; m <<= 1) {
            const float ov = __shfl_xor(minV[j], m, 32);
            const int oi = __shfl_xor(minI[j], m, 32);
            if (ov < minV[j] || (ov == minV[j] && oi < minI[j])) { // first-match ties
                minV[j] = ov;
                minI[j] = oi;
            }
        }
    }
    if (lane == 0 || lane == 16) {
        const int base = waveId * 16 + (lane >> 4) * 8;
#pragma unroll
        for (int j = 0; j < 8; ++j) idxTile[base + j] = minI[j];
    }
    __syncthreads();

    // ---- Phase 3: idx out, x_q gather/store (coalesced over rows), loss ----
    if (t < M_TILE) outIdx[rowBlock + t] = (float)idxTile[t];

    float acc2 = 0.f;
    {
        const int r = t & 63;
        const int ch = t >> 6;
        const int idxr = idxTile[r];
        const float* erow = emb + (size_t)idxr * LATENT_DIM;
        const float* xrow = x + (size_t)bIdx * LATENT_DIM * HW + hw0 + r;
        float* orow = outXq + (size_t)bIdx * LATENT_DIM * HW + hw0 + r;
        for (int cc = 0; cc < LATENT_DIM; cc += 2) {
            const int c = cc + ch;
            const float e = erow[c];
            const float xv = xrow[(size_t)c * HW];
            orow[(size_t)c * HW] = e; // straight-through == x_q numerically
            const float d = e - xv;
            acc2 += d * d;
        }
    }
    redBuf[t] = acc2;
    __syncthreads();
    for (int s = THREADS / 2; s > 0; s >>= 1) {
        if (t < s) redBuf[t] += redBuf[t + s];
        __syncthreads();
    }
    if (t == 0) lossPartial[blockIdx.x] = redBuf[0];
}

// ---------------------------------------------------------------------------
// Deterministic final loss reduction (no float atomics -> graph-replay stable).
// ---------------------------------------------------------------------------
__global__ __launch_bounds__(256) void vq_loss_reduce(const float* __restrict__ part,
                                                      float* __restrict__ lossSlot) {
    __shared__ float red[256];
    float a = 0.f;
    for (int i = threadIdx.x; i < N_ROWS_TOTAL / M_TILE; i += 256) a += part[i];
    red[threadIdx.x] = a;
    __syncthreads();
    for (int s = 128; s > 0; s >>= 1) {
        if (threadIdx.x < s) red[threadIdx.x] += red[threadIdx.x + s];
        __syncthreads();
    }
    if (threadIdx.x == 0)
        lossSlot[0] = red[0] * ((1.0f + BETA) / ((float)N_ROWS_TOTAL * (float)LATENT_DIM));
}

extern "C" void kernel_launch(void* const* d_in, const int* in_sizes, int n_in,
                              void* d_out, int out_size, void* d_ws, size_t ws_size,
                              hipStream_t stream) {
    const float* x = (const float*)d_in[0];   // [32,256,64,64]
    const float* emb = (const float*)d_in[1]; // [1024,256]

    float* outXq = (float*)d_out;                         // 33554432
    float* outIdx = outXq + (size_t)32 * LATENT_DIM * HW; // 131072
    float* lossSlot = outIdx + N_ROWS_TOTAL;              // 1

    unsigned short* packedB = (unsigned short*)d_ws; // 512 KB
    float* embNorm = (float*)((char*)d_ws +
                              (size_t)NUM_VECTORS * LATENT_DIM * sizeof(unsigned short));
    float* lossPartial = embNorm + NUM_VECTORS; // 2048 floats

    vq_prep<<<NUM_VECTORS, 256, 0, stream>>>(emb, packedB, embNorm);
    vq_main<<<N_ROWS_TOTAL / M_TILE, THREADS, 0, stream>>>(
        x, emb, packedB, embNorm, outXq, outIdx, lossPartial);
    vq_loss_reduce<<<1, 256, 0, stream>>>(lossPartial, lossSlot);
}
